// RNNSequence_12275016532277
// MI455X (gfx1250) — matrix operand order repaired and checked by reading, not compiled, not run on hardware
//
#include <hip/hip_runtime.h>
#include <stdint.h>

// Problem constants (match the reference).
#define BB 64
#define TT 1024
#define DD 256
#define HH 256

typedef __attribute__((ext_vector_type(16))) __bf16 v16bf;
typedef __attribute__((ext_vector_type(8)))  float  v8f;
typedef __attribute__((ext_vector_type(4)))  float  v4f;
typedef __attribute__((ext_vector_type(4)))  unsigned int v4u;
typedef __attribute__((ext_vector_type(4)))  unsigned short v4us;

union BF16x16 {
    v16bf v;
    v4u   q[2];
    unsigned int u[8];
    unsigned short s[16];
};

// --- fp32 -> bf16 conversion -----------------------------------------------
// Prefer the packed HW converter (1 VALU op per pair); bit-math RNE fallback.
#if __has_builtin(__builtin_amdgcn_cvt_pk_bf16_f32)
#define HAVE_PK_BF16 1
#else
#define HAVE_PK_BF16 0
#endif

__device__ __forceinline__ unsigned short f2bf_bits(float f) {
    unsigned u = __builtin_bit_cast(unsigned, f);
    u += 0x7FFFu + ((u >> 16) & 1u);
    return (unsigned short)(u >> 16);
}

// Pack two fp32 into one dword of bf16 (a -> [15:0], b -> [31:16]).
__device__ __forceinline__ unsigned cvt2bf(float a, float b) {
#if HAVE_PK_BF16
    auto p = __builtin_amdgcn_cvt_pk_bf16_f32(a, b);
    return __builtin_bit_cast(unsigned, p);
#else
    return (unsigned)f2bf_bits(a) | ((unsigned)f2bf_bits(b) << 16);
#endif
}

__device__ __forceinline__ unsigned short cvt1bf(float a) {
#if HAVE_PK_BF16
    return (unsigned short)(cvt2bf(a, a) & 0xFFFFu);
#else
    return f2bf_bits(a);
#endif
}

// Pack 16 contiguous fp32 (64B aligned) into a WMMA B-tile lane vector
// (B 32x16 as used here: each lane holds 16 contiguous K values, ascending).
__device__ __forceinline__ v16bf bf16_from_16f32(const float* __restrict__ p) {
    BF16x16 r;
    const v4f* q = (const v4f*)p;
    v4f c0 = q[0], c1 = q[1], c2 = q[2], c3 = q[3];
#pragma unroll
    for (int j = 0; j < 2; ++j) {
        r.u[j]     = cvt2bf(c0[2 * j], c0[2 * j + 1]);
        r.u[2 + j] = cvt2bf(c1[2 * j], c1[2 * j + 1]);
        r.u[4 + j] = cvt2bf(c2[2 * j], c2[2 * j + 1]);
        r.u[6 + j] = cvt2bf(c3[2 * j], c3[2 * j + 1]);
    }
    return r.v;
}

// Pack a WMMA A-tile lane vector (A 16x32 as used here: two adjacent 8-element
// K chunks: [k0, k0+8) then [k0+8, k0+16) for this half-wave).
__device__ __forceinline__ v16bf bf16_A_from_f32(const float* __restrict__ p0,
                                                 const float* __restrict__ p1) {
    BF16x16 r;
    const v4f* q0 = (const v4f*)p0;
    const v4f* q1 = (const v4f*)p1;
    v4f c0 = q0[0], c1 = q0[1], c2 = q1[0], c3 = q1[1];
#pragma unroll
    for (int j = 0; j < 2; ++j) {
        r.u[j]     = cvt2bf(c0[2 * j], c0[2 * j + 1]);
        r.u[2 + j] = cvt2bf(c1[2 * j], c1[2 * j + 1]);
        r.u[4 + j] = cvt2bf(c2[2 * j], c2[2 * j + 1]);
        r.u[6 + j] = cvt2bf(c3[2 * j], c3[2 * j + 1]);
    }
    return r.v;
}

// Branch-free tanh: sign(x) * (1 - e) / (1 + e), e = exp(-2|x|).
// Uses native v_exp_f32 / v_rcp_f32; well-behaved for all |x| (e -> 0).
__device__ __forceinline__ float fast_tanh(float x) {
    float ax = __builtin_fabsf(x);
    float e  = __expf(-2.0f * ax);
    float r  = (1.0f - e) * __builtin_amdgcn_rcpf(1.0f + e);
    return __builtin_copysignf(r, x);
}

// ---------------------------------------------------------------------------
// Kernel 0: one-time fp32 -> bf16 weight conversion into workspace.
// 65536 elements, 4 per thread -> 64 blocks x 256 threads.
// ---------------------------------------------------------------------------
__global__ __launch_bounds__(256)
void rnn_cvt_kernel(const float* __restrict__ src,
                    unsigned short* __restrict__ dst) {
    const int i = (blockIdx.x * 256 + threadIdx.x) * 4;
    v4f v = *(const v4f*)(src + i);
    unsigned lo = cvt2bf(v[0], v[1]);
    unsigned hi = cvt2bf(v[2], v[3]);
    unsigned long long o = (unsigned long long)lo | ((unsigned long long)hi << 32);
    *(unsigned long long*)(dst + i) = o;
}

// ---------------------------------------------------------------------------
// Kernel 1: x_proj = x @ W_ih^T + b_ih, written into out[b,t,:].
// Grid: 512 blocks x 256 threads. Each wave computes a 16-row x 256-col strip.
// PRE: B tiles read directly from the pre-converted bf16 weight copy (L2-hot,
// no per-tile VALU conversion). Fallback converts inline from fp32.
// ---------------------------------------------------------------------------
template <bool PRE>
__global__ __launch_bounds__(256)
void rnn_proj_kernel(const float* __restrict__ x,
                     const float* __restrict__ W_ih,
                     const unsigned short* __restrict__ Wbf,
                     const float* __restrict__ b_ih,
                     float* __restrict__ out) {
    const int lane = threadIdx.x & 31;
    const int wave = threadIdx.x >> 5;          // 0..7
    const int half = lane >> 4;                 // 0/1 (lane half)
    const int lm   = lane & 15;
    const int r0   = (blockIdx.x * 8 + wave) * 16;   // first of 16 rows (BT index)
    const int row  = r0 + lm;

    // Preload + convert this wave's A strip: 16 rows x 256 K (bf16), 8 K-blocks.
    // Conversion here is amortized over all 16 ntiles (the traffic-optimal spot).
    BF16x16 A[8];
    const float* xrow = x + (size_t)row * DD;
#pragma unroll
    for (int kb = 0; kb < 8; ++kb) {
        const float* p0 = xrow + kb * 32 + half * 16;
        const float* p1 = xrow + kb * 32 + half * 16 + 8;
        A[kb].v = bf16_A_from_f32(p0, p1);
    }

#pragma unroll 1
    for (int nt = 0; nt < 16; ++nt) {
        const int n = nt * 16 + lm;             // output column owned by lane
        const float bias = b_ih[n];
        v8f acc = {};
#pragma unroll
        for (int kb = 0; kb < 8; ++kb) {
            // B[k,n] = W_ih[n,k]; lane needs 16 contiguous K at kb*32+half*16.
            BF16x16 bt;
            if (PRE) {
                const v4u* p = (const v4u*)(Wbf + (size_t)n * DD + kb * 32 + half * 16);
                bt.q[0] = p[0];
                bt.q[1] = p[1];
            } else {
                bt.v = bf16_from_16f32(W_ih + (size_t)n * DD + kb * 32 + half * 16);
            }
            acc = __builtin_amdgcn_wmma_f32_16x16x32_bf16(
                false, A[kb].v, false, bt.v, (short)0, acc, false, false);
        }
        // C/D layout: VGPR i -> row r0 + i + 8*half, col n.
#pragma unroll
        for (int i = 0; i < 8; ++i) {
            out[(size_t)(r0 + i + 8 * half) * HH + n] = acc[i] + bias;
        }
    }
}

// ---------------------------------------------------------------------------
// Kernel 2: serial scan. 4 blocks (16 batch rows each) x 512 threads (16 waves,
// each wave owns a 16-column slice of h). W_hh^T tiles resident in VGPRs
// (converted once at setup), h double-buffered in LDS as bf16, xp streamed
// from `out` and overwritten in place with h. Per-step critical path:
// 16 ds_load_b128 (hoisted) -> 8 WMMAs over 4 accumulator chains (depth 2)
// -> branch-free tanh -> 8 ds_store_b16 -> 1 barrier.
// ---------------------------------------------------------------------------
__global__ __launch_bounds__(512)
void rnn_scan_kernel(const float* __restrict__ W_hh,
                     const float* __restrict__ b_hh,
                     float* __restrict__ out) {
    __shared__ unsigned short hbuf[2][16 * HH];     // 2 x 8 KB, bf16 h state

    const int lane = threadIdx.x & 31;
    const int wave = threadIdx.x >> 5;          // 0..15 -> n-tile
    const int half = lane >> 4;
    const int lm   = lane & 15;
    const int n    = wave * 16 + lm;            // h column owned by this lane
    const int b0   = blockIdx.x * 16;           // first batch row of this WG

    // Resident B tiles: B[k,n] = W_hh[n,k], 8 K-blocks of 32 (one-time setup).
    BF16x16 Bt[8];
#pragma unroll
    for (int kb = 0; kb < 8; ++kb) {
        const float* wrow = W_hh + (size_t)n * HH + kb * 32 + half * 16;
        Bt[kb].v = bf16_from_16f32(wrow);
    }
    const float bias = b_hh[n];

    // h0 = 0.
    for (int i = threadIdx.x; i < 16 * HH; i += 512) hbuf[0][i] = 0;

    // Global addresses of the 8 output elements this lane owns:
    // row b = b0 + i + 8*half (local M index), col n, time stride HH.
    size_t obase[8];
#pragma unroll
    for (int i = 0; i < 8; ++i)
        obase[i] = ((size_t)(b0 + i + 8 * half) * TT) * HH + n;

    __syncthreads();

#pragma unroll 1
    for (int t = 0; t < TT; ++t) {
        const unsigned short* hb = hbuf[t & 1];
        unsigned short*       hw = hbuf[(t + 1) & 1];

        // Stream xp for this step (phase-1 result, in place in `out`);
        // issued early so they overlap the WMMA chain. Prefetch t+1.
        float xp[8];
#pragma unroll
        for (int i = 0; i < 8; ++i) xp[i] = out[obase[i] + (size_t)t * HH];
        if (t + 1 < TT) {
#pragma unroll
            for (int i = 0; i < 8; ++i)
                __builtin_prefetch(&out[obase[i] + (size_t)(t + 1) * HH], 0, 1);
        }

        // Hoist all 16 LDS tile loads so dscnt waits are incremental.
        BF16x16 Aa[8];
#pragma unroll
        for (int kb = 0; kb < 8; ++kb) {
            const unsigned short* pa = hb + lm * HH + kb * 32 + half * 16;
            Aa[kb].q[0] = *(const v4u*)(pa);
            Aa[kb].q[1] = *(const v4u*)(pa + 8);
        }

        // h @ W_hh^T: 8 WMMAs over 4 independent accumulator chains (depth 2)
        // to minimize the serial WMMA latency on the recurrence path.
        v8f acc0 = {}, acc1 = {}, acc2 = {}, acc3 = {};
#pragma unroll
        for (int kb = 0; kb < 8; kb += 4) {
            acc0 = __builtin_amdgcn_wmma_f32_16x16x32_bf16(
                false, Aa[kb + 0].v, false, Bt[kb + 0].v, (short)0, acc0, false, false);
            acc1 = __builtin_amdgcn_wmma_f32_16x16x32_bf16(
                false, Aa[kb + 1].v, false, Bt[kb + 1].v, (short)0, acc1, false, false);
            acc2 = __builtin_amdgcn_wmma_f32_16x16x32_bf16(
                false, Aa[kb + 2].v, false, Bt[kb + 2].v, (short)0, acc2, false, false);
            acc3 = __builtin_amdgcn_wmma_f32_16x16x32_bf16(
                false, Aa[kb + 3].v, false, Bt[kb + 3].v, (short)0, acc3, false, false);
        }

        // h_new = tanh(recur + xp + b_hh); fp32 to out, bf16 to LDS for t+1.
#pragma unroll
        for (int i = 0; i < 8; ++i) {
            float s = ((acc0[i] + acc1[i]) + (acc2[i] + acc3[i])) + xp[i] + bias;
            float h = fast_tanh(s);
            out[obase[i] + (size_t)t * HH] = h;
            hw[(i + 8 * half) * HH + n] = cvt1bf(h);
        }
        __syncthreads();
    }
}

// ---------------------------------------------------------------------------
// Harness entry point.
// Inputs (setup_inputs order): x [B,T,D], W_ih [H,D], W_hh [H,H], b_ih, b_hh.
// d_out: [B,T,H] fp32.  d_ws: first 128 KB used for bf16 W_ih copy when large
// enough (branch depends only on ws_size -> deterministic).
// ---------------------------------------------------------------------------
extern "C" void kernel_launch(void* const* d_in, const int* in_sizes, int n_in,
                              void* d_out, int out_size, void* d_ws, size_t ws_size,
                              hipStream_t stream) {
    (void)in_sizes; (void)n_in; (void)out_size;
    const float* x    = (const float*)d_in[0];
    const float* W_ih = (const float*)d_in[1];
    const float* W_hh = (const float*)d_in[2];
    const float* b_ih = (const float*)d_in[3];
    const float* b_hh = (const float*)d_in[4];
    float* out = (float*)d_out;

    const size_t wbytes = (size_t)HH * DD * sizeof(unsigned short);  // 128 KB
    unsigned short* wbf = (unsigned short*)d_ws;

    if (ws_size >= wbytes) {
        // Phase 0: one-time W_ih fp32 -> bf16 (stays hot in 192 MB L2).
        rnn_cvt_kernel<<<dim3((HH * DD) / (256 * 4)), dim3(256), 0, stream>>>(W_ih, wbf);
        // Phase 1: x_proj for all (b,t) -> out, B tiles from bf16 copy.
        rnn_proj_kernel<true><<<dim3((BB * TT) / 128), dim3(256), 0, stream>>>(
            x, W_ih, wbf, b_ih, out);
    } else {
        // Fallback: inline weight conversion (no workspace needed).
        rnn_proj_kernel<false><<<dim3((BB * TT) / 128), dim3(256), 0, stream>>>(
            x, W_ih, wbf, b_ih, out);
    }
    // Phase 2: serial scan, batch-parallel across 4 workgroups.
    rnn_scan_kernel<<<dim3(BB / 16), dim3(512), 0, stream>>>(W_hh, b_hh, out);
}
